// RangeLinearQuantParamLayerWrapper_87814901334514
// MI455X (gfx1250) — 8-bit matrix operand order repaired and checked by reading, compiled, not run on hardware
//
#include <hip/hip_runtime.h>
#include <hip/hip_bf16.h>

typedef int v8i __attribute__((ext_vector_type(8)));

#define MDIM 4096
#define NDIM 4096
#define KDIM 4096
#define KD   (KDIM / 4)
#define BN   128
#define BM   128
#define RSTR 36
#define NST  (KDIM / 128)


__global__ void init_scalars(unsigned int* scal) {
    if (threadIdx.x < 4) scal[threadIdx.x] = 0u;
}

__global__ __launch_bounds__(256) void absmax_f32(const float* __restrict__ in,
                                                  int n4,
                                                  unsigned int* __restrict__ out) {
    const float4* in4 = (const float4*)in;
    float m = 0.0f;
    for (int i = blockIdx.x * blockDim.x + threadIdx.x; i < n4;
         i += gridDim.x * blockDim.x) {
        float4 v = in4[i];
        m = fmaxf(m, fmaxf(fmaxf(fabsf(v.x), fabsf(v.y)),
                           fmaxf(fabsf(v.z), fabsf(v.w))));
    }
    __shared__ float sm[256];
    sm[threadIdx.x] = m;
    __syncthreads();
    for (int s = 128; s > 0; s >>= 1) {
        if (threadIdx.x < s) sm[threadIdx.x] = fmaxf(sm[threadIdx.x], sm[threadIdx.x + s]);
        __syncthreads();
    }
    if (threadIdx.x == 0) atomicMax(out, __float_as_uint(sm[0]));
}

__device__ __forceinline__ int quant_one(float x, float scale) {
    int q = (int)rintf(x * scale);
    q = q < -128 ? -128 : (q > 127 ? 127 : q);
    return q;
}

__global__ __launch_bounds__(256) void quantize_i8(const float* __restrict__ in,
                                                   int n4,
                                                   const unsigned int* __restrict__ maxbits,
                                                   int* __restrict__ out) {
    const float m = __uint_as_float(*maxbits);
    const float scale = 255.0f / (2.0f * m);
    const float4* in4 = (const float4*)in;
    for (int i = blockIdx.x * blockDim.x + threadIdx.x; i < n4;
         i += gridDim.x * blockDim.x) {
        float4 v = in4[i];
        int q0 = quant_one(v.x, scale);
        int q1 = quant_one(v.y, scale);
        int q2 = quant_one(v.z, scale);
        int q3 = quant_one(v.w, scale);
        out[i] = (q0 & 255) | ((q1 & 255) << 8) | ((q2 & 255) << 16) | ((q3 & 255) << 24);
    }
}

__global__ void bias_requant(const float* __restrict__ b,
                             const unsigned int* __restrict__ scal,
                             int* __restrict__ bq, int n) {
    const float xmax = __uint_as_float(scal[0]);
    const float wmax = __uint_as_float(scal[1]);
    const float bmax = __uint_as_float(scal[2]);
    const float in_scale = 255.0f / (2.0f * xmax);
    const float w_scale  = 255.0f / (2.0f * wmax);
    const float b_scale  = 255.0f / (2.0f * bmax);
    const float accum_scale = in_scale * w_scale;
    int i = blockIdx.x * blockDim.x + threadIdx.x;
    if (i < n) {
        float base = rintf(b[i] * b_scale);
        base = fminf(fmaxf(base, -128.0f), 127.0f);
        float req = rintf(base * (accum_scale / b_scale));
        req = fminf(fmaxf(req, -2147483648.0f), 2147483520.0f);
        bq[i] = (int)req;
    }
}

__device__ __forceinline__ v8i load_a_frag_p(const int* __restrict__ p) {
    int2 p0 = *(const int2*)(p + 0);
    int2 p1 = *(const int2*)(p + 2);
    int2 p2 = *(const int2*)(p + 8);
    int2 p3 = *(const int2*)(p + 10);
    v8i a;
    a[0] = p0.x; a[1] = p0.y; a[2] = p1.x; a[3] = p1.y;
    a[4] = p2.x; a[5] = p2.y; a[6] = p3.x; a[7] = p3.y;
    return a;
}

__device__ __forceinline__ void async_copy_b128(const int* gsrc, unsigned ldsAddr) {
    asm volatile("global_load_async_to_lds_b128 %0, %1, off"
                 :: "v"(ldsAddr), "v"(gsrc));
}
__device__ __forceinline__ void wait_async0() {
    asm volatile("s_wait_asynccnt 0x0");
}

__global__ __launch_bounds__(256) void gemm_i8_wmma(const int* __restrict__ xq,
                                                    const int* __restrict__ wq,
                                                    const int* __restrict__ bq,
                                                    int* __restrict__ accum,
                                                    unsigned int* __restrict__ accum_absmax) {
    __shared__ __align__(16) int sB[2][BN * RSTR];

    const int lane = threadIdx.x & 31;
    const int wave = threadIdx.x >> 5;
    const int g    = lane >> 4;
    const int l15  = lane & 15;

    const int mBase = blockIdx.y * BM + wave * 16;
    const int nBase = blockIdx.x * BN;

    v8i acc[8];
#pragma unroll
    for (int t = 0; t < 8; ++t) {
        v8i z = {0, 0, 0, 0, 0, 0, 0, 0};
        acc[t] = z;
    }

    const int cr = threadIdx.x >> 1;
    const int ch = threadIdx.x & 1;
    const int* bnext = wq + (nBase + cr) * KD + ch * 16;
    const unsigned ldsW0 = (unsigned)(size_t)&sB[0][cr * RSTR + ch * 16];
    const unsigned ldsW1 = (unsigned)(size_t)&sB[1][cr * RSTR + ch * 16];
    const int* aP = xq + (mBase + l15) * KD + 4 * g;

    auto stage_copy = [&](const int* src, unsigned dst) {
        async_copy_b128(src + 0,  dst + 0);
        async_copy_b128(src + 4,  dst + 16);
        async_copy_b128(src + 8,  dst + 32);
        async_copy_b128(src + 12, dst + 48);
    };

    auto compute = [&](const int* __restrict__ sbuf, int kofs, v8i a) {
        const int fb = l15 * RSTR + kofs + 4 * g;
        int4 c0 = *(const int4*)(sbuf + fb);
        int4 c1 = *(const int4*)(sbuf + fb + 8);
#pragma unroll
        for (int t = 0; t < 8; ++t) {
            int4 n0 = c0, n1 = c1;
            if (t < 7) {
                const int nb = fb + (t + 1) * 16 * RSTR;
                n0 = *(const int4*)(sbuf + nb);
                n1 = *(const int4*)(sbuf + nb + 8);
            }
            v8i bf;
            bf[0] = c0.x; bf[1] = c0.y; bf[2] = c0.z; bf[3] = c0.w;
            bf[4] = c1.x; bf[5] = c1.y; bf[6] = c1.z; bf[7] = c1.w;
            acc[t] = __builtin_amdgcn_wmma_i32_16x16x64_iu8(
                          true, a,  true, bf, acc[t],
                          false,  false);
            c0 = n0; c1 = n1;
        }
    };

    stage_copy(bnext, ldsW0);
    bnext += 32;
    v8i aE0 = load_a_frag_p(aP);
    v8i aE1 = load_a_frag_p(aP + 16);
    aP += 32;
    wait_async0();
    __syncthreads();

    v8i aO0, aO1;
    for (int s = 0; s < NST; s += 2) {
        stage_copy(bnext, ldsW1);
        bnext += 32;
        aO0 = load_a_frag_p(aP);
        aO1 = load_a_frag_p(aP + 16);
        aP += 32;
        compute(sB[0], 0,  aE0);
        compute(sB[0], 16, aE1);
        wait_async0();
        __syncthreads();

        const bool more = (s + 2 < NST);
        if (more) {
            stage_copy(bnext, ldsW0);
            bnext += 32;
            aE0 = load_a_frag_p(aP);
            aE1 = load_a_frag_p(aP + 16);
            aP += 32;
        }
        compute(sB[1], 0,  aO0);
        compute(sB[1], 16, aO1);
        if (more) wait_async0();
        __syncthreads();
    }

    unsigned int lmax = 0u;
#pragma unroll
    for (int t = 0; t < 8; ++t) {
        const int n = nBase + t * 16 + l15;
        const int bias = bq[n];
#pragma unroll
        for (int r = 0; r < 8; ++r) {
            const int m = mBase + r + 8 * g;
            int v = acc[t][r] + bias;
            accum[m * NDIM + n] = v;
            unsigned int av = (unsigned int)(v < 0 ? -v : v);
            lmax = av > lmax ? av : lmax;
        }
    }
    __shared__ unsigned int smax[256];
    smax[threadIdx.x] = lmax;
    __syncthreads();
    for (int s = 128; s > 0; s >>= 1) {
        if (threadIdx.x < s) {
            unsigned int o = smax[threadIdx.x + s];
            if (o > smax[threadIdx.x]) smax[threadIdx.x] = o;
        }
        __syncthreads();
    }
    if (threadIdx.x == 0) atomicMax(accum_absmax, smax[0]);
}

__global__ __launch_bounds__(256) void requant_out(const int* __restrict__ accum,
                                                   int n4,
                                                   const unsigned int* __restrict__ scal,
                                                   float* __restrict__ out) {
    const float xmax = __uint_as_float(scal[0]);
    const float wmax = __uint_as_float(scal[1]);
    const float accum_scale = (255.0f / (2.0f * xmax)) * (255.0f / (2.0f * wmax));
    const float amax = (float)scal[3];
    const float ratio = 255.0f / (2.0f * amax);
    const float out_scale = ratio * accum_scale;
    const float inv_out_scale = 1.0f / out_scale;

    const int4* in4 = (const int4*)accum;
    float4* out4 = (float4*)out;
    for (int i = blockIdx.x * blockDim.x + threadIdx.x; i < n4;
         i += gridDim.x * blockDim.x) {
        int4 v = in4[i];
        float4 o;
        float q;
        q = rintf((float)v.x * ratio); q = fminf(fmaxf(q, -128.0f), 127.0f); o.x = q * inv_out_scale;
        q = rintf((float)v.y * ratio); q = fminf(fmaxf(q, -128.0f), 127.0f); o.y = q * inv_out_scale;
        q = rintf((float)v.z * ratio); q = fminf(fmaxf(q, -128.0f), 127.0f); o.z = q * inv_out_scale;
        q = rintf((float)v.w * ratio); q = fminf(fmaxf(q, -128.0f), 127.0f); o.w = q * inv_out_scale;
        out4[i] = o;
    }
}

extern "C" void kernel_launch(void* const* d_in, const int* in_sizes, int n_in,
                              void* d_out, int out_size, void* d_ws, size_t ws_size,
                              hipStream_t stream) {
    const float* x = (const float*)d_in[0];
    const float* W = (const float*)d_in[1];
    const float* b = (const float*)d_in[2];
    float* out = (float*)d_out;

    unsigned char* ws = (unsigned char*)d_ws;
    unsigned int* scal = (unsigned int*)ws;
    int* bq    = (int*)(ws + 256);
    int* xq    = (int*)(ws + 256 + 16384);
    int* wq    = (int*)(ws + 256 + 16384 + (16u << 20));
    int* accum = (int*)(ws + 256 + 16384 + (32u << 20));

    const int nElem  = MDIM * KDIM;
    const int nElem4 = nElem / 4;

    init_scalars<<<1, 32, 0, stream>>>(scal);

    absmax_f32<<<1024, 256, 0, stream>>>(x, nElem4, &scal[0]);
    absmax_f32<<<1024, 256, 0, stream>>>(W, nElem4, &scal[1]);
    absmax_f32<<<8, 256, 0, stream>>>(b, NDIM / 4, &scal[2]);

    quantize_i8<<<2048, 256, 0, stream>>>(x, nElem4, &scal[0], xq);
    quantize_i8<<<2048, 256, 0, stream>>>(W, nElem4, &scal[1], wq);
    bias_requant<<<NDIM / 256, 256, 0, stream>>>(b, scal, bq, NDIM);

    dim3 grid(NDIM / BN, MDIM / BM);
    gemm_i8_wmma<<<grid, 256, 0, stream>>>(xq, wq, bq, accum, &scal[3]);

    requant_out<<<4096, 256, 0, stream>>>(accum, nElem4, scal, out);
}
